// NeuralMemory_6365141532739
// MI455X (gfx1250) — matrix operand order repaired and checked by reading, not compiled, not run on hardware
//
#include <hip/hip_runtime.h>
#include <stdint.h>

// ---------------------------------------------------------------------------
// NeuralMemory step for MI455X (gfx1250, wave32, WMMA).
// bf16 operands (converted once), f32 accumulate via v_wmma_f32_16x16x32_bf16.
// Double-buffered LDS; async global->LDS copies (GLOBAL_LOAD_ASYNC_TO_LDS_B128,
// ASYNCcnt) for non-transposed tiles; packed-b32 transposed staging otherwise.
// ---------------------------------------------------------------------------

#define D_DIM  1024
#define NTOK   16384        // B*S = 4*4096
#define LRATE  0.01f
#define WDEC   0.01f
#define MAXALR 0.1f
#define EPS_A  1e-8f

typedef unsigned short u16;
typedef __attribute__((ext_vector_type(16))) __bf16 v16bf;
typedef __attribute__((ext_vector_type(8)))  float  v8f;
typedef int v4i_vs __attribute__((vector_size(16)));   // <4 x i32>, builtin param type

#if defined(__gfx1250__) && __has_builtin(__builtin_amdgcn_global_load_async_to_lds_b128)
#define USE_ASYNC 1
#else
#define USE_ASYNC 0
#endif

#if USE_ASYNC
static __device__ __forceinline__ void async_cp16(const u16* g, u16* l) {
  __builtin_amdgcn_global_load_async_to_lds_b128(
      (v4i_vs*)(g),
      (__attribute__((address_space(3))) v4i_vs*)(l), 0, 0);
}
#endif
static __device__ __forceinline__ void wait_async() {
#if USE_ASYNC
#if __has_builtin(__builtin_amdgcn_s_wait_asynccnt)
  __builtin_amdgcn_s_wait_asynccnt(0);
#else
  asm volatile("s_wait_asynccnt 0x0" ::: "memory");
#endif
#endif
}

static __device__ __forceinline__ u16 f2bfu(float f) {   // RNE f32 -> bf16 bits
  unsigned u = __builtin_bit_cast(unsigned, f);
  u += 0x7FFFu + ((u >> 16) & 1u);
  return (u16)(u >> 16);
}
static __device__ __forceinline__ float bfu2f(u16 h) {
  unsigned u = ((unsigned)h) << 16;
  return __builtin_bit_cast(float, u);
}

struct U128x2 { uint4 lo, hi; };

// A fragment: lane holds row M=lane&15; elems 0..7 -> K=lk..lk+7,
// elems 8..15 -> K=lk+16..lk+23, lk=8*(lane>>4). Two ds_load_b128.
static __device__ __forceinline__ v16bf ldfragA(const u16* p) {
  U128x2 t;
  t.lo = *(const uint4*)(p);
  t.hi = *(const uint4*)(p + 16);
  return __builtin_bit_cast(v16bf, t);
}
// B fragment: lane holds col N=lane&15; elems 0..7 -> K=kb..kb+7,
// elems 8..15 -> K=kb+16..kb+23, kb=8*(lane>>4).
static __device__ __forceinline__ v16bf ldfragB(const u16* p) {
  U128x2 t;
  t.lo = *(const uint4*)(p);
  t.hi = *(const uint4*)(p + 16);
  return __builtin_bit_cast(v16bf, t);
}

constexpr int BM = 128, BN = 128, BK = 32;
constexpr int LDSS = 40;   // u16 stride: 80B rows keep 16B alignment, spread banks

static __device__ __forceinline__ void mma_tile(const u16* As, const u16* Bs,
                                                int lane, int wm, int wn,
                                                v8f acc[2][4]) {
  const int lrow = lane & 15;
  const int lka  = (lane >> 4) * 8;
  const int lkb  = (lane >> 4) * 16;
  v16bf af[2], bfr[4];
#pragma unroll
  for (int i = 0; i < 2; ++i)
    af[i] = ldfragA(&As[(wm + i * 16 + lrow) * LDSS + lka]);
#pragma unroll
  for (int j = 0; j < 4; ++j)
    bfr[j] = ldfragB(&Bs[(wn + j * 16 + lrow) * LDSS + lka]);
#pragma unroll
  for (int i = 0; i < 2; ++i)
#pragma unroll
    for (int j = 0; j < 4; ++j)
      acc[i][j] = __builtin_amdgcn_wmma_f32_16x16x32_bf16(
          false, af[i], false, bfr[j], (short)0, acc[i][j], false, false);
}

// ---------------------------------------------------------------------------
// GEMM: C = A[M,K](bf16) @ B(bf16) (+bias) with fused epilogues.
//   BT=false: B is [K,N].  BT=true: B is [N,K] (computes A @ B^T).
//   EPI 0: v = acc+bias              -> out0 / out0b
//   EPI 1: z = acc+bias -> out0; h = resb + silu(z) -> out1 / out1b
//   EPI 2: v = (acc + addend) * dsilu(zld) -> out0 / out0b   (no bias)
// ---------------------------------------------------------------------------
template <bool BT, int EPI>
__global__ __launch_bounds__(256) void gemm_bf16(
    const u16* __restrict__ A, const u16* __restrict__ B,
    const float* __restrict__ bias, const float* __restrict__ addend,
    const u16* __restrict__ resb, const float* __restrict__ zld,
    float* __restrict__ out0, u16* __restrict__ out0b,
    float* __restrict__ out1, u16* __restrict__ out1b,
    int M, int N, int K) {
  __shared__ u16 As[2][BM * LDSS];
  __shared__ u16 Bs[2][BN * LDSS];

  const int tid  = threadIdx.x;
  const int lane = tid & 31;
  const int wid  = tid >> 5;
  const int wm   = (wid >> 1) * 32;
  const int wn   = (wid & 1) * 64;
  const int m0   = blockIdx.y * BM;
  const int n0   = blockIdx.x * BN;

  // staging coords: A rows arow, arow+64, 8 halves at ac8.
  const int arow = tid >> 2, ac8 = (tid & 3) * 8;
  // !BT B staging: adjacent k rows bk2, bk2+1 at cols bn8..bn8+7 -> packed b32 stores
  const int bk2 = (tid >> 4) * 2, bn8 = (tid & 15) * 8;
  const u16* pa = A + (size_t)(m0 + arow) * K + ac8;
  const u16* pb = BT ? (B + (size_t)(n0 + arow) * K + ac8)
                     : (B + (size_t)bk2 * N + n0 + bn8);

  uint4 ra0, ra1, rb0, rb1;
  auto tile_prefetch = [&](u16* as, u16* bs) {
#if USE_ASYNC
    async_cp16(pa, &as[arow * LDSS + ac8]);
    async_cp16(pa + (size_t)64 * K, &as[(arow + 64) * LDSS + ac8]);
    pa += BK;
    if (BT) {
      async_cp16(pb, &bs[arow * LDSS + ac8]);
      async_cp16(pb + (size_t)64 * K, &bs[(arow + 64) * LDSS + ac8]);
      pb += BK;
    } else {
      rb0 = *(const uint4*)(pb);
      rb1 = *(const uint4*)(pb + (size_t)N);
      pb += (size_t)BK * N;
    }
#else
    ra0 = *(const uint4*)(pa);
    ra1 = *(const uint4*)(pa + (size_t)64 * K);
    pa += BK;
    if (BT) {
      rb0 = *(const uint4*)(pb);
      rb1 = *(const uint4*)(pb + (size_t)64 * K);
      pb += BK;
    } else {
      rb0 = *(const uint4*)(pb);
      rb1 = *(const uint4*)(pb + (size_t)N);
      pb += (size_t)BK * N;
    }
#endif
    (void)as; (void)bs;
  };
  auto tile_commit = [&](u16* as, u16* bs) {
#if !USE_ASYNC
    *(uint4*)&as[arow * LDSS + ac8]        = ra0;
    *(uint4*)&as[(arow + 64) * LDSS + ac8] = ra1;
    if (BT) {
      *(uint4*)&bs[arow * LDSS + ac8]        = rb0;
      *(uint4*)&bs[(arow + 64) * LDSS + ac8] = rb1;
    }
#endif
    if (!BT) {   // transpose [K,N] tile -> Bs[n][k], k-pairs packed as b32
      const u16* s0 = (const u16*)&rb0;
      const u16* s1 = (const u16*)&rb1;
#pragma unroll
      for (int e = 0; e < 8; ++e) {
        unsigned v = (unsigned)s0[e] | ((unsigned)s1[e] << 16);
        *(unsigned*)&bs[(bn8 + e) * LDSS + bk2] = v;
      }
    }
    wait_async();
    (void)as;
  };

  v8f acc[2][4] = {};
  const int nk = K / BK;
  tile_prefetch(As[0], Bs[0]);
  tile_commit(As[0], Bs[0]);
  __syncthreads();
  for (int kt = 0; kt < nk; ++kt) {
    const int cur = kt & 1;
    const bool more = (kt + 1) < nk;
    if (more) tile_prefetch(As[cur ^ 1], Bs[cur ^ 1]);
    mma_tile(As[cur], Bs[cur], lane, wm, wn, acc);
    if (more) tile_commit(As[cur ^ 1], Bs[cur ^ 1]);
    __syncthreads();
  }

#pragma unroll
  for (int i = 0; i < 2; ++i)
#pragma unroll
    for (int j = 0; j < 4; ++j)
#pragma unroll
      for (int r = 0; r < 8; ++r) {
        int m = m0 + wm + i * 16 + r + ((lane >> 4) << 3);
        int n = n0 + wn + j * 16 + (lane & 15);
        size_t idx = (size_t)m * N + n;
        float v = acc[i][j][r];
        if (EPI != 2 && bias) v += bias[n];
        if (EPI == 0) {
          if (out0)  out0[idx]  = v;
          if (out0b) out0b[idx] = f2bfu(v);
        } else if (EPI == 1) {
          if (out0) out0[idx] = v;
          float s = 1.f / (1.f + __expf(-v));
          float h = bfu2f(resb[idx]) + v * s;         // res + silu(z)
          if (out1)  out1[idx]  = h;
          if (out1b) out1b[idx] = f2bfu(h);
        } else {                                      // EPI == 2
          v += addend[idx];
          float z = zld[idx];
          float s = 1.f / (1.f + __expf(-z));
          float r2 = v * (s * (1.f + z * (1.f - s))); // * dsilu(z)
          if (out0)  out0[idx]  = r2;
          if (out0b) out0b[idx] = f2bfu(r2);
        }
      }
}

// ---------------------------------------------------------------------------
// Weight-grad GEMM: C[M,N] += A^T @ G, A:[Ktok,M] bf16, G:[Ktok,N] bf16.
// Split-K over blockIdx.z, f32 global atomic accumulation (C pre-zeroed).
// Both tiles transposed in LDS with packed-b32 stores (adjacent token rows).
// ---------------------------------------------------------------------------
__global__ __launch_bounds__(256) void gemm_at_bf16(
    const u16* __restrict__ A, const u16* __restrict__ G,
    float* __restrict__ C, int Mdim, int Ndim, int kChunk) {
  __shared__ u16 As[2][BM * LDSS];
  __shared__ u16 Bs[2][BN * LDSS];

  const int tid  = threadIdx.x;
  const int lane = tid & 31;
  const int wid  = tid >> 5;
  const int wm   = (wid >> 1) * 32;
  const int wn   = (wid & 1) * 64;
  const int m0   = blockIdx.y * BM;
  const int n0   = blockIdx.x * BN;
  const int kc0  = blockIdx.z * kChunk;

  const int tk2 = (tid >> 4) * 2, tc8 = (tid & 15) * 8;  // adjacent token rows
  const u16* pa = A + (size_t)(kc0 + tk2) * Mdim + m0 + tc8;
  const u16* pg = G + (size_t)(kc0 + tk2) * Ndim + n0 + tc8;

  uint4 ra0, ra1, rb0, rb1;
  auto tile_prefetch = [&]() {
    ra0 = *(const uint4*)(pa);
    ra1 = *(const uint4*)(pa + (size_t)Mdim);
    rb0 = *(const uint4*)(pg);
    rb1 = *(const uint4*)(pg + (size_t)Ndim);
    pa += (size_t)BK * Mdim;
    pg += (size_t)BK * Ndim;
  };
  auto tile_commit = [&](u16* as, u16* bs) {
    const u16* a0 = (const u16*)&ra0; const u16* a1 = (const u16*)&ra1;
    const u16* b0 = (const u16*)&rb0; const u16* b1 = (const u16*)&rb1;
#pragma unroll
    for (int e = 0; e < 8; ++e) {
      *(unsigned*)&as[(tc8 + e) * LDSS + tk2] =
          (unsigned)a0[e] | ((unsigned)a1[e] << 16);
      *(unsigned*)&bs[(tc8 + e) * LDSS + tk2] =
          (unsigned)b0[e] | ((unsigned)b1[e] << 16);
    }
  };

  v8f acc[2][4] = {};
  const int nk = kChunk / BK;
  tile_prefetch();
  tile_commit(As[0], Bs[0]);
  __syncthreads();
  for (int kt = 0; kt < nk; ++kt) {
    const int cur = kt & 1;
    const bool more = (kt + 1) < nk;
    if (more) tile_prefetch();
    mma_tile(As[cur], Bs[cur], lane, wm, wn, acc);
    if (more) tile_commit(As[cur ^ 1], Bs[cur ^ 1]);
    __syncthreads();
  }

#pragma unroll
  for (int i = 0; i < 2; ++i)
#pragma unroll
    for (int j = 0; j < 4; ++j)
#pragma unroll
      for (int r = 0; r < 8; ++r) {
        int m = m0 + wm + i * 16 + r + ((lane >> 4) << 3);
        int n = n0 + wn + j * 16 + (lane & 15);
        atomicAdd(&C[(size_t)m * Ndim + n], acc[i][j][r]);
      }
}

// ---------------------------------------------------------------------------
// Small kernels
// ---------------------------------------------------------------------------
__global__ void zero_kernel(float* __restrict__ p, long long n) {
  long long i = (long long)blockIdx.x * blockDim.x + threadIdx.x;
  long long s = (long long)gridDim.x * blockDim.x;
  for (; i < n; i += s) p[i] = 0.f;
}

__global__ void cvt_bf16_kernel(const float* __restrict__ s, u16* __restrict__ d,
                                long long n4) {
  long long i = (long long)blockIdx.x * blockDim.x + threadIdx.x;
  long long st = (long long)gridDim.x * blockDim.x;
  for (; i < n4; i += st) {
    float4 v = ((const float4*)s)[i];
    uint2 o;
    o.x = (unsigned)f2bfu(v.x) | ((unsigned)f2bfu(v.y) << 16);
    o.y = (unsigned)f2bfu(v.z) | ((unsigned)f2bfu(v.w) << 16);
    ((uint2*)d)[i] = o;
  }
}

// one wave per token: alr = MAX_ALR * sigmoid(x . Wlr + blr)
__global__ __launch_bounds__(256) void alr_kernel(
    const float* __restrict__ x, const float* __restrict__ Wlr,
    const float* __restrict__ blr, float* __restrict__ alr) {
  int wid = threadIdx.x >> 5, lane = threadIdx.x & 31;
  int t = blockIdx.x * 8 + wid;
  const float* xr = x + (size_t)t * D_DIM;
  float s = 0.f;
  for (int d = lane; d < D_DIM; d += 32) s += xr[d] * Wlr[d];
#pragma unroll
  for (int off = 16; off > 0; off >>= 1) s += __shfl_xor(s, off, 32);
  if (lane == 0) alr[t] = MAXALR / (1.f + __expf(-(s + blr[0])));
}

// g2out = 2*alr/D*(pred - vf) (f32);  dz2b = bf16(g2out * dsilu(z2))
__global__ void grad_out_kernel(
    const float* __restrict__ pred, const float* __restrict__ vf,
    const float* __restrict__ z2, const float* __restrict__ alr,
    float* __restrict__ g2out, u16* __restrict__ dz2b, long long n) {
  long long i = (long long)blockIdx.x * blockDim.x + threadIdx.x;
  long long st = (long long)gridDim.x * blockDim.x;
  for (; i < n; i += st) {
    int t = (int)(i >> 10);  // i / D_DIM
    float g = (2.0f / (float)D_DIM) * alr[t] * (pred[i] - vf[i]);
    g2out[i] = g;
    float z = z2[i];
    float s = 1.f / (1.f + __expf(-z));
    dz2b[i] = f2bfu(g * (s * (1.f + z * (1.f - s))));
  }
}

// bias grad from bf16 grads: gb[j] += sum over row chunk of G[t, j]
__global__ void colsum_kernel(const u16* __restrict__ G, float* __restrict__ gb,
                              int rowsPerChunk) {
  int j = blockIdx.x * blockDim.x + threadIdx.x;
  int r0 = blockIdx.y * rowsPerChunk;
  float s = 0.f;
  for (int t = 0; t < rowsPerChunk; ++t) s += bfu2f(G[(size_t)(r0 + t) * D_DIM + j]);
  atomicAdd(&gb[j], s);
}

// AdamW at t=1 from zero state; emits surprise = -g. pn / pnb nullable.
__global__ void adamw_kernel(const float* __restrict__ p, const float* __restrict__ g,
                             float* __restrict__ pn, u16* __restrict__ pnb,
                             float* __restrict__ surp, long long n) {
  long long i = (long long)blockIdx.x * blockDim.x + threadIdx.x;
  long long st = (long long)gridDim.x * blockDim.x;
  for (; i < n; i += st) {
    float gi = g[i];
    float v = p[i] * (1.f - LRATE * WDEC) - LRATE * gi / (fabsf(gi) + EPS_A);
    if (pn)  pn[i]  = v;
    if (pnb) pnb[i] = f2bfu(v);
    surp[i] = -gi;
  }
}

// ---------------------------------------------------------------------------
extern "C" void kernel_launch(void* const* d_in, const int* in_sizes, int n_in,
                              void* d_out, int out_size, void* d_ws, size_t ws_size,
                              hipStream_t stream) {
  (void)in_sizes; (void)n_in; (void)out_size; (void)ws_size;

  const float* x   = (const float*)d_in[0];
  const float* Wk  = (const float*)d_in[1];
  const float* bk  = (const float*)d_in[2];
  const float* Wq  = (const float*)d_in[3];
  const float* bq  = (const float*)d_in[4];
  const float* Wv  = (const float*)d_in[5];
  const float* bv  = (const float*)d_in[6];
  const float* Wlr = (const float*)d_in[7];
  const float* blr = (const float*)d_in[8];
  const float* W1  = (const float*)d_in[9];
  const float* b1  = (const float*)d_in[10];
  const float* W2  = (const float*)d_in[11];
  const float* b2  = (const float*)d_in[12];

  const long long ND = (long long)NTOK * D_DIM;   // 16,777,216
  const long long DD = (long long)D_DIM * D_DIM;  // 1,048,576

  float* w = (float*)d_ws;
  float* vf    = w + 0 * ND;
  float* z1    = w + 1 * ND;
  float* z2    = w + 2 * ND;
  float* pred  = w + 3 * ND;
  float* g2out = w + 4 * ND;
  u16* ub   = (u16*)(w + 5 * ND);
  u16* xb   = ub + 0 * ND;
  u16* kfb  = ub + 1 * ND;
  u16* qfb  = ub + 2 * ND;
  u16* h1b  = ub + 3 * ND;
  u16* dz1b = ub + 4 * ND;
  u16* dz2b = xb;    // x bf16 dead after projections
  u16* h1qb = kfb;   // kf bf16 dead after gW1
  u16* Wkb  = ub + 5 * ND;
  u16* Wqb  = Wkb + DD;
  u16* Wvb  = Wqb + DD;
  u16* W1b  = Wvb + DD;
  u16* W2b  = W1b + DD;
  u16* W1nb = W2b + DD;
  u16* W2nb = W1nb + DD;
  float* sf  = w + 5 * ND + (5 * ND + 7 * DD) / 2;
  float* gW1 = sf;            float* gW2 = sf + DD;
  float* gb1 = sf + 2 * DD;   float* gb2 = gb1 + 1024;
  float* b1n = gb1 + 2048;    float* b2n = b1n + 1024;
  float* alr = b2n + 1024;

  float* out       = (float*)d_out;
  float* retrieved = out;
  float* sW1 = out + ND;
  float* sb1 = sW1 + DD;
  float* sW2 = sb1 + 1024;
  float* sb2 = sW2 + DD;

  dim3 blk(256);
  dim3 gTok(D_DIM / BN, NTOK / BM);        // (8, 128)
  dim3 gWgt(D_DIM / BN, D_DIM / BM, 8);    // split-K weight grads

  cvt_bf16_kernel<<<4096, blk, 0, stream>>>(x,  xb,  ND / 4);
  cvt_bf16_kernel<<<1024, blk, 0, stream>>>(Wk, Wkb, DD / 4);
  cvt_bf16_kernel<<<1024, blk, 0, stream>>>(Wq, Wqb, DD / 4);
  cvt_bf16_kernel<<<1024, blk, 0, stream>>>(Wv, Wvb, DD / 4);
  cvt_bf16_kernel<<<1024, blk, 0, stream>>>(W1, W1b, DD / 4);
  cvt_bf16_kernel<<<1024, blk, 0, stream>>>(W2, W2b, DD / 4);

  zero_kernel<<<512, blk, 0, stream>>>(gW1, 2 * DD + 2048);
  alr_kernel<<<NTOK / 8, blk, 0, stream>>>(x, Wlr, blr, alr);

  gemm_bf16<false, 0><<<gTok, blk, 0, stream>>>(xb, Wkb, bk, nullptr, nullptr, nullptr,
                                                nullptr, kfb, nullptr, nullptr,
                                                NTOK, D_DIM, D_DIM);
  gemm_bf16<false, 0><<<gTok, blk, 0, stream>>>(xb, Wqb, bq, nullptr, nullptr, nullptr,
                                                nullptr, qfb, nullptr, nullptr,
                                                NTOK, D_DIM, D_DIM);
  gemm_bf16<false, 0><<<gTok, blk, 0, stream>>>(xb, Wvb, bv, nullptr, nullptr, nullptr,
                                                vf, nullptr, nullptr, nullptr,
                                                NTOK, D_DIM, D_DIM);

  gemm_bf16<false, 1><<<gTok, blk, 0, stream>>>(kfb, W1b, b1, nullptr, kfb, nullptr,
                                                z1, nullptr, nullptr, h1b,
                                                NTOK, D_DIM, D_DIM);
  gemm_bf16<false, 1><<<gTok, blk, 0, stream>>>(h1b, W2b, b2, nullptr, h1b, nullptr,
                                                z2, nullptr, pred, nullptr,
                                                NTOK, D_DIM, D_DIM);

  grad_out_kernel<<<4096, blk, 0, stream>>>(pred, vf, z2, alr, g2out, dz2b, ND);

  gemm_at_bf16<<<gWgt, blk, 0, stream>>>(h1b, dz2b, gW2, D_DIM, D_DIM, NTOK / 8);
  colsum_kernel<<<dim3(D_DIM / 256, 32), blk, 0, stream>>>(dz2b, gb2, NTOK / 32);

  gemm_bf16<true, 2><<<gTok, blk, 0, stream>>>(dz2b, W2b, nullptr, g2out, nullptr, z1,
                                               nullptr, dz1b, nullptr, nullptr,
                                               NTOK, D_DIM, D_DIM);

  gemm_at_bf16<<<gWgt, blk, 0, stream>>>(kfb, dz1b, gW1, D_DIM, D_DIM, NTOK / 8);
  colsum_kernel<<<dim3(D_DIM / 256, 32), blk, 0, stream>>>(dz1b, gb1, NTOK / 32);

  adamw_kernel<<<1024, blk, 0, stream>>>(W1, gW1, nullptr, W1nb, sW1, DD);
  adamw_kernel<<<4,    blk, 0, stream>>>(b1, gb1, b1n, nullptr, sb1, 1024);
  adamw_kernel<<<1024, blk, 0, stream>>>(W2, gW2, nullptr, W2nb, sW2, DD);
  adamw_kernel<<<4,    blk, 0, stream>>>(b2, gb2, b2n, nullptr, sb2, 1024);

  gemm_bf16<false, 1><<<gTok, blk, 0, stream>>>(qfb, W1nb, b1n, nullptr, qfb, nullptr,
                                                nullptr, nullptr, nullptr, h1qb,
                                                NTOK, D_DIM, D_DIM);
  gemm_bf16<false, 1><<<gTok, blk, 0, stream>>>(h1qb, W2nb, b2n, nullptr, h1qb, nullptr,
                                                nullptr, nullptr, retrieved, nullptr,
                                                NTOK, D_DIM, D_DIM);
}
